// ContextualAttention_Enhance_44169443672106
// MI455X (gfx1250) — matrix operand order repaired and checked by reading, not compiled, not run on hardware
//
#include <hip/hip_runtime.h>
#include <hip/hip_bf16.h>

typedef _Float16 half_t;
typedef __attribute__((ext_vector_type(16))) _Float16 v16h;
typedef __attribute__((ext_vector_type(8)))  float    v8f;
typedef __attribute__((ext_vector_type(2)))  float    v2f;

// ---- problem constants ----
constexpr int Bn   = 2;
constexpr int CIN  = 64;
constexpr int CI   = 16;
constexpr int Hh   = 96;
constexpr int Wd   = 96;
constexpr int HW   = Hh * Wd;          // 9216
constexpr int PSz  = 7;
constexpr int P2   = 3;
constexpr int WSz  = 21;
constexpr int W2c  = 10;
constexpr int NCand = WSz * WSz;       // 441
constexpr int TopK = 100;
constexpr int S0   = 4;
constexpr int NHq  = 24;
constexpr int NWq  = 24;
constexpr int Qn   = NHq * NWq;        // 576 queries / batch
constexpr float SCL = 10.0f;
constexpr int PD    = CI * PSz * PSz;  // 784 patch-feature dim
constexpr int PDpad = 800;             // padded to 25 chunks of 32
constexpr int REG   = 33;              // candidate union region per 4x4 query tile
constexpr int NREG  = REG * REG;       // 1089
constexpr int NPAD  = 1104;            // 69 tiles of 16
constexpr int NCT   = 69;
constexpr int QT    = 6;               // 6x6 query tiles of 4x4 per batch
constexpr int NQT   = QT * QT;         // 36
constexpr int RP    = 39;              // phi halo region = 33 + 2*3

__device__ __forceinline__ int clampi(int v, int lo, int hi) {
  return v < lo ? lo : (v > hi ? hi : v);
}

__device__ __forceinline__ v8f wmma_f32(v2f a, v2f b, v8f c) {
  // V_WMMA_F32_16X16X4_F32 : exact fp32 matrix op
  return __builtin_amdgcn_wmma_f32_16x16x4_f32(false, a, false, b, (short)0, c,
                                               false, false);
}
__device__ __forceinline__ v8f wmma_f16(v16h a, v16h b, v8f c) {
  // V_WMMA_F32_16X16X32_F16 : throughput path for the score GEMM
  return __builtin_amdgcn_wmma_f32_16x16x32_f16(false, a, false, b, (short)0, c,
                                                false, false);
}

// gfx1250 async gather into LDS (ASYNCcnt-tracked, no VGPR data movement)
__device__ __forceinline__ void async_ld_b32(unsigned lds_off, const void* g) {
  asm volatile("global_load_async_to_lds_b32 %0, %1, off"
               :: "v"(lds_off), "v"((unsigned long long)(uintptr_t)g)
               : "memory");
}
__device__ __forceinline__ void async_ld_b128(unsigned lds_off, const void* g) {
  asm volatile("global_load_async_to_lds_b128 %0, %1, off"
               :: "v"(lds_off), "v"((unsigned long long)(uintptr_t)g)
               : "memory");
}
__device__ __forceinline__ void wait_async0() {
  asm volatile("s_wait_asynccnt 0x0" ::: "memory");
}

// ------------------------------------------------------------------
// Kernel 0: zero the accumulation buffers (harness poisons ws once)
// ------------------------------------------------------------------
__global__ void k_zero(float* __restrict__ imgAcc, float* __restrict__ cnt) {
  int i = blockIdx.x * blockDim.x + threadIdx.x;
  int nA = Bn * CI * HW;
  int nC = Bn * HW;
  if (i < nA) imgAcc[i] = 0.0f;
  if (i < nC) cnt[i] = 0.0f;
}

// ------------------------------------------------------------------
// Kernel 1: fused theta/phi/g 1x1 conv.  WMMA f32 16x16x4.
//   A = W (16x64 rows=out-chan), B = vid tile (64 x 16 pixels)
//   theta/phi stored as [b][pix][16] f16, g as [b][pix][16] f32.
// grid (Bn*576, 3) block 32 (1 wave)
// ------------------------------------------------------------------
__global__ void k_qkv(const float* __restrict__ vid,
                      const float* __restrict__ wth, const float* __restrict__ bth,
                      const float* __restrict__ wph, const float* __restrict__ bph,
                      const float* __restrict__ wg,  const float* __restrict__ bg,
                      half_t* __restrict__ thetaH, half_t* __restrict__ phiH,
                      float* __restrict__ gF) {
  int bx = blockIdx.x;
  int b  = bx / (HW / 16);
  int pt = bx % (HW / 16);
  int which = blockIdx.y;           // 0=theta 1=phi 2=g
  const float* w  = which == 0 ? wth : (which == 1 ? wph : wg);
  const float* bi = which == 0 ? bth : (which == 1 ? bph : bg);

  int lane = threadIdx.x & 31;
  int m    = lane & 15;
  int hi   = (lane >> 4) & 1;       // lanes 16..31
  int pix0 = pt * 16;

  v8f acc = {0.f,0.f,0.f,0.f,0.f,0.f,0.f,0.f};
  const float* vb = vid + (size_t)b * CIN * HW;
#pragma unroll 4
  for (int k0 = 0; k0 < CIN; k0 += 4) {
    v2f a, bm;
    int ka0 = k0 + (hi ? 2 : 0);
    int ka1 = k0 + (hi ? 3 : 1);
    a.x = w[m * CIN + ka0];
    a.y = w[m * CIN + ka1];
    bm.x = vb[(size_t)ka0 * HW + pix0 + m];
    bm.y = vb[(size_t)ka1 * HW + pix0 + m];
    acc = wmma_f32(a, bm, acc);
  }
#pragma unroll
  for (int r = 0; r < 8; ++r) {
    int oc  = r + (hi ? 8 : 0);
    int pix = pix0 + m;
    float v = acc[r] + bi[oc];
    size_t idx = ((size_t)b * HW + pix) * CI + oc;
    if (which == 0)      thetaH[idx] = (half_t)v;
    else if (which == 1) phiH[idx]   = (half_t)v;
    else                 gF[idx]     = v;
  }
}

// ------------------------------------------------------------------
// Kernel 2a: attention scores.  For each 4x4 query tile, compute the
// dense 16 x 1104 score block against its 33x33 candidate union
// region with WMMA f32_16x16x32_f16 (K = 800, 25 chunks).
//  - Qp tile (16 q x 800, f16) gathered into LDS with
//    global_load_async_to_lds_b32 (per-feature-pair gather).
//  - phi halo region (39x39 x 16ch f16) staged into LDS with
//    global_load_async_to_lds_b128; all B operands are then LDS reads.
// grid (Bn*36, 3) block 128 (4 waves); wave-slot strides candidate tiles.
// ------------------------------------------------------------------
__global__ __launch_bounds__(128)
void k_scores(const half_t* __restrict__ thetaH,
              const half_t* __restrict__ phiH,
              float* __restrict__ S) {
  __shared__ __align__(16) half_t ldsQ[16][PDpad];     // 25.6 KB
  __shared__ __align__(16) half_t ldsP[RP * RP][CI];   // 48.7 KB

  int bx  = blockIdx.x;
  int b   = bx / NQT;
  int qt  = bx % NQT;
  int qty = qt / QT, qtx = qt % QT;
  int py0 = qty * 16, px0 = qtx * 16;   // query-pixel origin of tile
  int by0 = py0 - W2c, bx0 = px0 - W2c; // candidate region origin (can be <0)
  int gy0 = by0 - P2,  gx0 = bx0 - P2;  // phi halo region origin

  int tid = threadIdx.x;
  const half_t* thb = thetaH + (size_t)b * HW * CI;
  const half_t* phb = phiH + (size_t)b * HW * CI;

  // ---- async gather: Qp tile (pairs of adjacent channels = b32) ----
  for (int idx = tid; idx < 16 * (PD / 2); idx += blockDim.x) {
    int q  = idx / (PD / 2);
    int f  = (idx % (PD / 2)) * 2;
    int o = f >> 4, c = f & 15;
    int qy = qty * 4 + (q >> 2);
    int qx = qtx * 4 + (q & 3);
    int Y = clampi(qy * S0 + o / PSz - P2, 0, Hh - 1);
    int X = clampi(qx * S0 + o % PSz - P2, 0, Wd - 1);
    async_ld_b32((unsigned)(uintptr_t)&ldsQ[q][f],
                 thb + (size_t)(Y * Wd + X) * CI + c);
  }
  // K-padding region of ldsQ (f in [784,800)) is zero
  for (int idx = tid; idx < 16 * (PDpad - PD); idx += blockDim.x) {
    int q = idx / (PDpad - PD);
    int f = PD + idx % (PDpad - PD);
    ldsQ[q][f] = (half_t)0.f;
  }
  // ---- async stage: phi halo region, 2 x b128 per region pixel ----
  for (int idx = tid; idx < RP * RP * 2; idx += blockDim.x) {
    int pix = idx >> 1, hh = idx & 1;
    int yy = pix / RP, xx = pix % RP;
    int Y = clampi(gy0 + yy, 0, Hh - 1);
    int X = clampi(gx0 + xx, 0, Wd - 1);
    async_ld_b128((unsigned)(uintptr_t)&ldsP[pix][hh * 8],
                  phb + (size_t)(Y * Wd + X) * CI + hh * 8);
  }
  wait_async0();
  __syncthreads();

  int wave = tid >> 5;
  int lane = tid & 31;
  int n    = lane & 15;
  int hi   = (lane >> 4) & 1;
  int slot = blockIdx.y * 4 + wave;   // 12 wave-slots over 69 ctiles

  float* Sq = S + (size_t)(b * NQT + qt) * 16 * NPAD;

  for (int ct = slot; ct < NCT; ct += 12) {
    int r  = ct * 16 + n;
    int rr = r < NREG ? r : NREG - 1;        // padded lanes: deterministic clamp
    int ry = rr / REG, rx = rr % REG;
    int ci = clampi(by0 + ry, 0, Hh - 1);
    int cj = clampi(bx0 + rx, 0, Wd - 1);

    v8f acc = {0.f,0.f,0.f,0.f,0.f,0.f,0.f,0.f};
#pragma unroll 5
    for (int t = 0; t < PDpad / 32; ++t) {
      v16h a, bm;
      // A operand: 16-bit A 16x32 layout (VGPR v<4: K=2v(+8 hi); v>=4: 16+2(v-4)(+8 hi))
#pragma unroll
      for (int v = 0; v < 8; ++v) {
        int klo = (v < 4) ? (2 * v + (hi ? 8 : 0))
                          : (16 + 2 * (v - 4) + (hi ? 8 : 0));
        int f = t * 32 + klo;
        a[2 * v]     = ldsQ[n][f];
        a[2 * v + 1] = ldsQ[n][f + 1];
      }
      // B operand: 16-bit B 32x16 layout (VGPR v<4: K=2v(+8 hi); v>=4: 16+2(v-4)(+8 hi)); LDS halo reads
#pragma unroll
      for (int v = 0; v < 8; ++v) {
        int klo = (v < 4) ? (2 * v + (hi ? 8 : 0))
                          : (16 + 2 * (v - 4) + (hi ? 8 : 0));
        int f = t * 32 + klo;
        half_t b0 = (half_t)0.f, b1 = (half_t)0.f;
        if (f < PD) {
          int o = f >> 4, c = f & 15;
          int yy = clampi(ci + o / PSz - P2, 0, Hh - 1) - gy0;  // provably in [0,38]
          int xx = clampi(cj + o % PSz - P2, 0, Wd - 1) - gx0;
          const half_t* p = &ldsP[yy * RP + xx][c];
          b0 = p[0];
          b1 = p[1];
        }
        bm[2 * v]     = b0;
        bm[2 * v + 1] = b1;
      }
      acc = wmma_f16(a, bm, acc);
    }
    // D: VGPR r -> query row r (+8 for hi lanes), col = n
#pragma unroll
    for (int rr8 = 0; rr8 < 8; ++rr8) {
      int qrow = rr8 + (hi ? 8 : 0);
      Sq[(size_t)qrow * NPAD + ct * 16 + n] = acc[rr8];
    }
  }
}

// ------------------------------------------------------------------
// Kernel 2b: per query -- gather 441 scores, top-100 (wave32 shuffle
// argmax), softmax, weighted patch sum from g, overlap-add scatter
// with global f32 atomics; also accumulate coverage counts.
// grid (Bn*576) block 32
// ------------------------------------------------------------------
__global__ void k_topk(const float* __restrict__ S,
                       const float* __restrict__ gF,
                       float* __restrict__ imgAcc,
                       float* __restrict__ cnt) {
  __shared__ float topw[TopK];
  __shared__ int   topc[TopK];
  __shared__ int   sci[TopK];
  __shared__ int   scj[TopK];

  int bq = blockIdx.x;
  int b  = bq / Qn;
  int q  = bq % Qn;
  int qy = q / NWq, qx = q % NWq;
  int qi = qy * S0, qj = qx * S0;
  int qt = (qy / 4) * QT + (qx / 4);
  int qq = (qy & 3) * 4 + (qx & 3);
  int tyq = qy & 3, txq = qx & 3;
  int lane = threadIdx.x & 31;

  const float* Srow = S + ((size_t)(b * NQT + qt) * 16 + qq) * NPAD;
  const float* gb = gF + (size_t)b * HW * CI;
  __builtin_prefetch(Srow, 0, 3);                 // global_prefetch_b8
  __builtin_prefetch(gb + (size_t)(qi * Wd + qj) * CI, 0, 3);

  float val[14];
#pragma unroll
  for (int s = 0; s < 14; ++s) {
    int c = s * 32 + lane;
    float v = -3.4e38f;
    if (c < NCand) {
      int dy = c / WSz, dx = c % WSz;
      int r = (tyq * 4 + dy) * REG + (txq * 4 + dx);
      v = Srow[r];
    }
    val[s] = v;
  }

  // ---- iterative wave-wide arg-max, 100 rounds ----
  for (int k = 0; k < TopK; ++k) {
    float bm = -3.4e38f;
    int bi = 0x7fffffff;
#pragma unroll
    for (int s = 0; s < 14; ++s) {
      if (val[s] > bm) { bm = val[s]; bi = s * 32 + lane; }
    }
#pragma unroll
    for (int off = 16; off >= 1; off >>= 1) {
      float ov = __shfl_xor(bm, off, 32);
      int   oi = __shfl_xor(bi, off, 32);
      if (ov > bm || (ov == bm && oi < bi)) { bm = ov; bi = oi; }
    }
    if (lane == (bi & 31)) val[bi >> 5] = -3.4e38f;
    if (lane == 0) { topw[k] = bm; topc[k] = bi; }
  }
  __syncthreads();

  // ---- softmax over the 100 selected scores ----
  float vmax = topw[0];
  float psum = 0.f;
  for (int k = lane; k < TopK; k += 32) {
    float e = __expf(SCL * (topw[k] - vmax));
    topw[k] = e;
    psum += e;
  }
#pragma unroll
  for (int off = 16; off >= 1; off >>= 1) psum += __shfl_xor(psum, off, 32);
  float inv = 1.0f / psum;
  __syncthreads();
  for (int k = lane; k < TopK; k += 32) {
    topw[k] *= inv;
    int cl = topc[k];
    sci[k] = clampi(qi + cl / WSz - W2c, 0, Hh - 1);
    scj[k] = clampi(qj + cl % WSz - W2c, 0, Wd - 1);
  }
  __syncthreads();

  // ---- coverage counts (once per query, per patch position) ----
  for (int t = lane; t < PSz * PSz; t += 32) {
    int Y = clampi(qi + t / PSz - P2, 0, Hh - 1);
    int X = clampi(qj + t % PSz - P2, 0, Wd - 1);
    atomicAdd(&cnt[(size_t)b * HW + Y * Wd + X], 1.0f);
  }

  // ---- weighted patch sum + overlap-add scatter ----
  for (int f = lane; f < PD; f += 32) {
    int o = f >> 4, c = f & 15;
    int dy = o / PSz, dx = o % PSz;
    float acc = 0.f;
#pragma unroll 4
    for (int k = 0; k < TopK; ++k) {
      int Y = clampi(sci[k] + dy - P2, 0, Hh - 1);
      int X = clampi(scj[k] + dx - P2, 0, Wd - 1);
      acc += topw[k] * gb[(size_t)(Y * Wd + X) * CI + c];
    }
    int Yq = clampi(qi + dy - P2, 0, Hh - 1);
    int Xq = clampi(qj + dx - P2, 0, Wd - 1);
    atomicAdd(&imgAcc[((size_t)b * CI + c) * HW + Yq * Wd + Xq], acc);
  }
}

// ------------------------------------------------------------------
// Kernel 2c: normalize overlap-add -> z[b][c][p]
// ------------------------------------------------------------------
__global__ void k_norm(const float* __restrict__ imgAcc,
                       const float* __restrict__ cnt,
                       float* __restrict__ zbuf) {
  int i = blockIdx.x * blockDim.x + threadIdx.x;
  if (i >= Bn * HW) return;
  int b = i / HW, p = i % HW;
  float inv = 1.0f / fmaxf(cnt[i], 1.0f);
#pragma unroll
  for (int c = 0; c < CI; ++c) {
    size_t idx = ((size_t)b * CI + c) * HW + p;
    zbuf[idx] = imgAcc[idx] * inv;
  }
}

// ------------------------------------------------------------------
// Kernel 3a: fold y = W_w z + b_w into the final conv:
//   Wcomb = Wc33[:, :64] @ W_w  (64x16),  bcomb = b_c33 + Wc33[:, :64] @ b_w
// ------------------------------------------------------------------
__global__ void k_comb(const float* __restrict__ w_w, const float* __restrict__ b_w,
                       const float* __restrict__ w_c33, const float* __restrict__ b_c33,
                       float* __restrict__ Wcomb, float* __restrict__ bcomb) {
  int oc = threadIdx.x;
  if (oc >= CIN) return;
  for (int i = 0; i < CI; ++i) {
    float s = 0.f;
    for (int c = 0; c < CIN; ++c) s += w_c33[oc * (2 * CIN) + c] * w_w[c * CI + i];
    Wcomb[oc * CI + i] = s;
  }
  float sb = b_c33[oc];
  for (int c = 0; c < CIN; ++c) sb += w_c33[oc * (2 * CIN) + c] * b_w[c];
  bcomb[oc] = sb;
}

// ------------------------------------------------------------------
// Kernel 3b: final GEMM  out[64, p] = Wcomb @ z (K=16) + Wc33[:,64:] @ vid
//            (K=64) + bcomb.  WMMA f32 16x16x4, K=80 total.
// grid (Bn*576, 4) block 32
// ------------------------------------------------------------------
__global__ void k_final(const float* __restrict__ zbuf,
                        const float* __restrict__ vid,
                        const float* __restrict__ Wcomb,
                        const float* __restrict__ bcomb,
                        const float* __restrict__ w_c33,
                        float* __restrict__ out) {
  int bx = blockIdx.x;
  int b  = bx / (HW / 16);
  int pt = bx % (HW / 16);
  int oct = blockIdx.y;
  int lane = threadIdx.x & 31;
  int m    = lane & 15;
  int hi   = (lane >> 4) & 1;
  int pix  = pt * 16 + m;

  const float* zb = zbuf + (size_t)b * CI * HW;
  const float* vb = vid + (size_t)b * CIN * HW;

  v8f acc = {0.f,0.f,0.f,0.f,0.f,0.f,0.f,0.f};
#pragma unroll 4
  for (int k0 = 0; k0 < 80; k0 += 4) {
    int ka0 = k0 + (hi ? 2 : 0);
    int ka1 = k0 + (hi ? 3 : 1);
    int oc  = oct * 16 + m;
    v2f a, bm;
    a.x = (ka0 < CI) ? Wcomb[oc * CI + ka0] : w_c33[oc * (2 * CIN) + CIN + (ka0 - CI)];
    a.y = (ka1 < CI) ? Wcomb[oc * CI + ka1] : w_c33[oc * (2 * CIN) + CIN + (ka1 - CI)];
    bm.x = (ka0 < CI) ? zb[(size_t)ka0 * HW + pix] : vb[(size_t)(ka0 - CI) * HW + pix];
    bm.y = (ka1 < CI) ? zb[(size_t)ka1 * HW + pix] : vb[(size_t)(ka1 - CI) * HW + pix];
    acc = wmma_f32(a, bm, acc);
  }
#pragma unroll
  for (int r = 0; r < 8; ++r) {
    int oc = oct * 16 + r + (hi ? 8 : 0);
    out[((size_t)b * CIN + oc) * HW + pix] = acc[r] + bcomb[oc];
  }
}

// ------------------------------------------------------------------
extern "C" void kernel_launch(void* const* d_in, const int* in_sizes, int n_in,
                              void* d_out, int out_size, void* d_ws, size_t ws_size,
                              hipStream_t stream) {
  const float* vid    = (const float*)d_in[0];
  const float* w_th   = (const float*)d_in[1];
  const float* b_th   = (const float*)d_in[2];
  const float* w_ph   = (const float*)d_in[3];
  const float* b_ph   = (const float*)d_in[4];
  const float* w_g    = (const float*)d_in[5];
  const float* b_g    = (const float*)d_in[6];
  const float* w_w    = (const float*)d_in[7];
  const float* b_w    = (const float*)d_in[8];
  const float* w_c33  = (const float*)d_in[9];
  const float* b_c33  = (const float*)d_in[10];
  float* out = (float*)d_out;

  // workspace carve-out (256B aligned slabs)
  char* ws = (char*)d_ws;
  size_t off = 0;
  auto take = [&](size_t bytes) {
    char* p = ws + off;
    off += (bytes + 255) & ~(size_t)255;
    return p;
  };
  half_t* thetaH = (half_t*)take((size_t)Bn * HW * CI * sizeof(half_t));
  half_t* phiH   = (half_t*)take((size_t)Bn * HW * CI * sizeof(half_t));
  float*  gF     = (float*) take((size_t)Bn * HW * CI * sizeof(float));
  float*  S      = (float*) take((size_t)Bn * NQT * 16 * NPAD * sizeof(float));
  float*  imgAcc = (float*) take((size_t)Bn * CI * HW * sizeof(float));
  float*  cnt    = (float*) take((size_t)Bn * HW * sizeof(float));
  float*  zbuf   = (float*) take((size_t)Bn * CI * HW * sizeof(float));
  float*  Wcomb  = (float*) take((size_t)CIN * CI * sizeof(float));
  float*  bcomb  = (float*) take((size_t)CIN * sizeof(float));
  (void)ws_size; (void)in_sizes; (void)n_in; (void)out_size;

  {
    int n = Bn * CI * HW;
    k_zero<<<(n + 255) / 256, 256, 0, stream>>>(imgAcc, cnt);
  }
  k_qkv<<<dim3(Bn * (HW / 16), 3), 32, 0, stream>>>(
      vid, w_th, b_th, w_ph, b_ph, w_g, b_g, thetaH, phiH, gF);
  k_scores<<<dim3(Bn * NQT, 3), 128, 0, stream>>>(thetaH, phiH, S);
  k_topk<<<Bn * Qn, 32, 0, stream>>>(S, gF, imgAcc, cnt);
  {
    int n = Bn * HW;
    k_norm<<<(n + 255) / 256, 256, 0, stream>>>(imgAcc, cnt, zbuf);
  }
  k_comb<<<1, 64, 0, stream>>>(w_w, b_w, w_c33, b_c33, Wcomb, bcomb);
  k_final<<<dim3(Bn * (HW / 16), 4), 32, 0, stream>>>(
      zbuf, vid, Wcomb, bcomb, w_c33, out);
}
